// DecoderLayer_61357902791157
// MI455X (gfx1250) — compile-verified, matrix operand order checked by reading, not run on hardware
//
#include <hip/hip_runtime.h>
#include <stdint.h>
#include <stddef.h>

#define DM      1024
#define DI      2048
#define NDSTATE 16
#define DCONVW  4
#define DTR     64
#define DFF     4096
#define LSEQ    2048
#define XZW     (2*DI)
#define DBLW    (DTR + 2*NDSTATE)

typedef __attribute__((ext_vector_type(16))) __bf16 v16bf;
typedef __attribute__((ext_vector_type(8)))  float  v8f;
typedef __attribute__((ext_vector_type(8)))  int    v8i;

union FragBF { v16bf v; unsigned int u[8]; };
union FragI8 { v8i   v; unsigned int u[8]; };

#if defined(__has_builtin)
#  if __has_builtin(__builtin_amdgcn_global_load_async_to_lds_b128) && \
      __has_builtin(__builtin_amdgcn_s_wait_asynccnt)
#    define HAVE_ASYNC_LDS 1
#  endif
#endif
#ifndef HAVE_ASYNC_LDS
#  define HAVE_ASYNC_LDS 0
#endif

#if HAVE_ASYNC_LDS
typedef int v4i_async __attribute__((vector_size(16)));
typedef __attribute__((address_space(1))) v4i_async* g_v4i_p;
typedef __attribute__((address_space(3))) v4i_async* l_v4i_p;
__device__ __forceinline__ void async_cp16(const void* g, void* l) {
  __builtin_amdgcn_global_load_async_to_lds_b128((g_v4i_p)g, (l_v4i_p)l, 0, 0);
}
__device__ __forceinline__ void async_wait0() { __builtin_amdgcn_s_wait_asynccnt(0); }
#endif

__device__ __forceinline__ unsigned short f2bf(float f) {
  union { float f; unsigned u; } c; c.f = f;
  return (unsigned short)((c.u + 0x7FFFu + ((c.u >> 16) & 1u)) >> 16);
}
__device__ __forceinline__ float sigmoidf_(float x) { return 1.f / (1.f + __expf(-x)); }
__device__ __forceinline__ float siluf_(float x)    { return x * sigmoidf_(x); }
__device__ __forceinline__ float softplusf_(float x){ return (x > 20.f) ? x : log1pf(__expf(x)); }
__device__ __forceinline__ float geluf_(float x) {
  float x3 = x * x * x;
  return 0.5f * x * (1.f + tanhf(0.7978845608028654f * (x + 0.044715f * x3)));
}

__device__ __forceinline__ float blkSum(float v, float* s) {
  int t = threadIdx.x; s[t] = v; __syncthreads();
  for (int k = 128; k > 0; k >>= 1) { if (t < k) s[t] += s[t + k]; __syncthreads(); }
  float r = s[0]; __syncthreads(); return r;
}
__device__ __forceinline__ float blkMax(float v, float* s) {
  int t = threadIdx.x; s[t] = v; __syncthreads();
  for (int k = 128; k > 0; k >>= 1) { if (t < k) s[t] = fmaxf(s[t], s[t + k]); __syncthreads(); }
  float r = s[0]; __syncthreads(); return r;
}

__global__ void k_cvt_bf16(const float* __restrict__ in, unsigned short* __restrict__ out, int n) {
  int i = blockIdx.x * blockDim.x + threadIdx.x;
  if (i < n) out[i] = f2bf(in[i]);
}

#define BM   64
#define BN   128
#define BKH  32
#define SSTR 40

enum { EPI_NONE = 0, EPI_STORE_BF16 = 1, EPI_SOFTPLUS_BIAS = 2 };

template<int EPI>
__global__ __launch_bounds__(256)
void k_gemm_bf16(const unsigned short* __restrict__ A, int lda,
                 const unsigned short* __restrict__ B, int ldb,
                 float* __restrict__ C, int ldc,
                 unsigned short* __restrict__ Cbf,
                 const float* __restrict__ bias,
                 int M, int N, int K)
{
  const int tid  = threadIdx.x;
  const int lane = tid & 31;
  const int wave = tid >> 5;
  const int wm = wave >> 2, wn = wave & 3;
  const int l16 = lane & 15, half = lane >> 4;
  const int m0 = blockIdx.y * BM;
  const int n0 = blockIdx.x * BN;

  v8f acc[2][2] = {};

  const int ar = tid >> 2, ac = (tid & 3) * 8;
  const int br = tid >> 1, bc = (tid & 1) * 16;

#if HAVE_ASYNC_LDS
  __shared__ __align__(16) unsigned short sA[2][BM * SSTR];
  __shared__ __align__(16) unsigned short sB[2][BN * SSTR];
  const bool aOK = (m0 + ar) < M;
  const bool bOK = (n0 + br) < N;
  {
    uint4 z = make_uint4(0u, 0u, 0u, 0u);
    if (!aOK) { *(uint4*)&sA[0][ar*SSTR+ac] = z; *(uint4*)&sA[1][ar*SSTR+ac] = z; }
    if (!bOK) {
      uint4* q0 = (uint4*)&sB[0][br*SSTR+bc]; q0[0] = z; q0[1] = z;
      uint4* q1 = (uint4*)&sB[1][br*SSTR+bc]; q1[0] = z; q1[1] = z;
    }
  }
  const unsigned short* gptrA = A + (size_t)(m0 + ar) * lda + ac;
  const unsigned short* gptrB = B + (size_t)(n0 + br) * ldb + bc;
  if (aOK) async_cp16(gptrA, &sA[0][ar*SSTR+ac]);
  if (bOK) { async_cp16(gptrB,     &sB[0][br*SSTR+bc]);
             async_cp16(gptrB + 8, &sB[0][br*SSTR+bc+8]); }
  async_wait0();
  __syncthreads();
#else
  __shared__ __align__(16) unsigned short sA1[BM * SSTR];
  __shared__ __align__(16) unsigned short sB1[BN * SSTR];
#endif

  for (int k0 = 0; k0 < K; k0 += BKH) {
#if HAVE_ASYNC_LDS
    const int cur = (k0 / BKH) & 1;
    if (k0 + BKH < K) {
      if (aOK) async_cp16(gptrA + k0 + BKH, &sA[cur^1][ar*SSTR+ac]);
      if (bOK) { async_cp16(gptrB + k0 + BKH,     &sB[cur^1][br*SSTR+bc]);
                 async_cp16(gptrB + k0 + BKH + 8, &sB[cur^1][br*SSTR+bc+8]); }
    }
    const unsigned short* pA = sA[cur];
    const unsigned short* pB = sB[cur];
#else
    if (k0 + BKH < K) {
      if (m0 + ar < M) __builtin_prefetch(A + (size_t)(m0 + ar) * lda + (k0 + BKH) + ac, 0, 1);
      if (n0 + br < N) __builtin_prefetch(B + (size_t)(n0 + br) * ldb + (k0 + BKH) + bc, 0, 1);
    }
    {
      uint4 v = make_uint4(0u, 0u, 0u, 0u);
      int gm = m0 + ar;
      if (gm < M) v = *(const uint4*)(A + (size_t)gm * lda + k0 + ac);
      *(uint4*)&sA1[ar * SSTR + ac] = v;
    }
    {
      uint4 v0 = make_uint4(0u, 0u, 0u, 0u), v1 = v0;
      int gn = n0 + br;
      if (gn < N) {
        const uint4* p = (const uint4*)(B + (size_t)gn * ldb + k0 + bc);
        v0 = p[0]; v1 = p[1];
      }
      uint4* q = (uint4*)&sB1[br * SSTR + bc];
      q[0] = v0; q[1] = v1;
    }
    __syncthreads();
    const unsigned short* pA = sA1;
    const unsigned short* pB = sB1;
#endif

    FragBF fa[2], fb[2];
#pragma unroll
    for (int s2 = 0; s2 < 2; ++s2) {
      int mr = (wm * 32 + s2 * 16 + l16) * SSTR;
      int nr = (wn * 32 + s2 * 16 + l16) * SSTR;
#pragma unroll
      for (int v = 0; v < 8; ++v) {
        int kb = ((v >> 2) << 4) + half * 8 + ((v & 3) << 1);
        fa[s2].u[v] = *(const unsigned int*)&pA[mr + kb];
        fb[s2].u[v] = *(const unsigned int*)&pB[nr + kb];
      }
    }
#pragma unroll
    for (int i = 0; i < 2; ++i)
#pragma unroll
      for (int j = 0; j < 2; ++j)
        acc[i][j] = __builtin_amdgcn_wmma_f32_16x16x32_bf16(
            false, fa[i].v, false, fb[j].v, (short)0, acc[i][j], false, false);

#if HAVE_ASYNC_LDS
    async_wait0();
    __syncthreads();
#else
    __syncthreads();
#endif
  }

#pragma unroll
  for (int i = 0; i < 2; ++i) {
#pragma unroll
    for (int j = 0; j < 2; ++j) {
      int n = n0 + wn * 32 + j * 16 + l16;
      if (n >= N) continue;
#pragma unroll
      for (int r = 0; r < 8; ++r) {
        int m = m0 + wm * 32 + i * 16 + half * 8 + r;
        if (m >= M) continue;
        float val = acc[i][j][r];
        if (EPI == EPI_SOFTPLUS_BIAS) val = softplusf_(val + bias[n]);
        C[(size_t)m * ldc + n] = val;
        if (EPI == EPI_STORE_BF16) Cbf[(size_t)m * ldc + n] = f2bf(val);
      }
    }
  }
}

#define BKI   64
#define SSTRB 80

template<int DOGELU>
__global__ __launch_bounds__(256)
void k_gemm_i8(const signed char* __restrict__ A, int lda,
               const signed char* __restrict__ B, int ldb,
               const float* __restrict__ rsA,
               const float* __restrict__ gB,
               const float* __restrict__ bias,
               float* __restrict__ C, int ldc,
               int M, int N, int K)
{
  const int tid  = threadIdx.x;
  const int lane = tid & 31;
  const int wave = tid >> 5;
  const int wm = wave >> 2, wn = wave & 3;
  const int l16 = lane & 15, half = lane >> 4;
  const int m0 = blockIdx.y * BM;
  const int n0 = blockIdx.x * BN;

  v8i acc[2][2] = {};

  const int ar = tid >> 2, ac = (tid & 3) * 16;
  const int br = tid >> 1, bc = (tid & 1) * 32;

#if HAVE_ASYNC_LDS
  __shared__ __align__(16) signed char sA[2][BM * SSTRB];
  __shared__ __align__(16) signed char sB[2][BN * SSTRB];
  const bool aOK = (m0 + ar) < M;
  const bool bOK = (n0 + br) < N;
  {
    uint4 z = make_uint4(0u, 0u, 0u, 0u);
    if (!aOK) { *(uint4*)&sA[0][ar*SSTRB+ac] = z; *(uint4*)&sA[1][ar*SSTRB+ac] = z; }
    if (!bOK) {
      uint4* q0 = (uint4*)&sB[0][br*SSTRB+bc]; q0[0] = z; q0[1] = z;
      uint4* q1 = (uint4*)&sB[1][br*SSTRB+bc]; q1[0] = z; q1[1] = z;
    }
  }
  const signed char* gptrA = A + (size_t)(m0 + ar) * lda + ac;
  const signed char* gptrB = B + (size_t)(n0 + br) * ldb + bc;
  if (aOK) async_cp16(gptrA, &sA[0][ar*SSTRB+ac]);
  if (bOK) { async_cp16(gptrB,      &sB[0][br*SSTRB+bc]);
             async_cp16(gptrB + 16, &sB[0][br*SSTRB+bc+16]); }
  async_wait0();
  __syncthreads();
#else
  __shared__ __align__(16) signed char sA1[BM * SSTRB];
  __shared__ __align__(16) signed char sB1[BN * SSTRB];
#endif

  for (int k0 = 0; k0 < K; k0 += BKI) {
#if HAVE_ASYNC_LDS
    const int cur = (k0 / BKI) & 1;
    if (k0 + BKI < K) {
      if (aOK) async_cp16(gptrA + k0 + BKI, &sA[cur^1][ar*SSTRB+ac]);
      if (bOK) { async_cp16(gptrB + k0 + BKI,      &sB[cur^1][br*SSTRB+bc]);
                 async_cp16(gptrB + k0 + BKI + 16, &sB[cur^1][br*SSTRB+bc+16]); }
    }
    const signed char* pA = sA[cur];
    const signed char* pB = sB[cur];
#else
    if (k0 + BKI < K) {
      if (m0 + ar < M) __builtin_prefetch(A + (size_t)(m0 + ar) * lda + (k0 + BKI) + ac, 0, 1);
      if (n0 + br < N) __builtin_prefetch(B + (size_t)(n0 + br) * ldb + (k0 + BKI) + bc, 0, 1);
    }
    {
      uint4 v = make_uint4(0u, 0u, 0u, 0u);
      int gm = m0 + ar;
      if (gm < M) v = *(const uint4*)(A + (size_t)gm * lda + k0 + ac);
      *(uint4*)&sA1[ar * SSTRB + ac] = v;
    }
    {
      uint4 v0 = make_uint4(0u, 0u, 0u, 0u), v1 = v0;
      int gn = n0 + br;
      if (gn < N) {
        const uint4* p = (const uint4*)(B + (size_t)gn * ldb + k0 + bc);
        v0 = p[0]; v1 = p[1];
      }
      uint4* q = (uint4*)&sB1[br * SSTRB + bc];
      q[0] = v0; q[1] = v1;
    }
    __syncthreads();
    const signed char* pA = sA1;
    const signed char* pB = sB1;
#endif

    FragI8 fa[2], fb[2];
#pragma unroll
    for (int s2 = 0; s2 < 2; ++s2) {
      int mr = (wm * 32 + s2 * 16 + l16) * SSTRB;
      int nr = (wn * 32 + s2 * 16 + l16) * SSTRB;
#pragma unroll
      for (int v = 0; v < 8; ++v) {
        int kb = ((v >> 1) << 4) + half * 8 + ((v & 1) << 2);
        fa[s2].u[v] = *(const unsigned int*)&pA[mr + kb];
        fb[s2].u[v] = *(const unsigned int*)&pB[nr + kb];
      }
    }
#pragma unroll
    for (int i = 0; i < 2; ++i)
#pragma unroll
      for (int j = 0; j < 2; ++j)
        acc[i][j] = __builtin_amdgcn_wmma_i32_16x16x64_iu8(
            true, fa[i].v, true, fb[j].v, acc[i][j], false, false);

#if HAVE_ASYNC_LDS
    async_wait0();
    __syncthreads();
#else
    __syncthreads();
#endif
  }

  const float ga = gB[0];
#pragma unroll
  for (int i = 0; i < 2; ++i) {
#pragma unroll
    for (int j = 0; j < 2; ++j) {
      int n = n0 + wn * 32 + j * 16 + l16;
      if (n >= N) continue;
#pragma unroll
      for (int r = 0; r < 8; ++r) {
        int m = m0 + wm * 32 + i * 16 + half * 8 + r;
        if (m >= M) continue;
        float val = (float)acc[i][j][r] * (ga * rsA[m]) + bias[n];
        if (DOGELU) val = geluf_(val);
        C[(size_t)m * ldc + n] = val;
      }
    }
  }
}

__global__ __launch_bounds__(256)
void k_conv_silu(const float* __restrict__ xz, const float* __restrict__ cw,
                 const float* __restrict__ cb, float* __restrict__ u,
                 unsigned short* __restrict__ ubf)
{
  int idx = blockIdx.x * 256 + threadIdx.x;
  if (idx >= LSEQ * DI) return;
  int d = idx & (DI - 1);
  int l = idx >> 11;
  float acc = cb[d];
#pragma unroll
  for (int t = 0; t < DCONVW; ++t) {
    int ll = l - (DCONVW - 1) + t;
    if (ll >= 0) acc += cw[d * DCONVW + t] * xz[(size_t)ll * XZW + d];
  }
  float s = siluf_(acc);
  u[idx] = s;
  ubf[idx] = f2bf(s);
}

__global__ __launch_bounds__(256)
void k_scan(const float* __restrict__ delta, const float* __restrict__ u,
            const float* __restrict__ dbl,  const float* __restrict__ A_log,
            const float* __restrict__ Dp,   const float* __restrict__ xz,
            unsigned short* __restrict__ ybf)
{
  int d = blockIdx.x * 256 + threadIdx.x;
  if (d >= DI) return;
  float Ar[NDSTATE], h[NDSTATE];
#pragma unroll
  for (int n = 0; n < NDSTATE; ++n) { Ar[n] = -__expf(A_log[d * NDSTATE + n]); h[n] = 0.f; }
  const float Dpd = Dp[d];
  for (int l = 0; l < LSEQ; ++l) {
    float dl = delta[(size_t)l * DI + d];
    float ul = u[(size_t)l * DI + d];
    float du = dl * ul;
    const float* bcrow = dbl + (size_t)l * DBLW;
    float y = 0.f;
#pragma unroll
    for (int n = 0; n < NDSTATE; ++n) {
      h[n] = __expf(dl * Ar[n]) * h[n] + du * bcrow[DTR + n];
      y = fmaf(h[n], bcrow[DTR + NDSTATE + n], y);
    }
    y = fmaf(ul, Dpd, y);
    float z = xz[(size_t)l * XZW + DI + d];
    y *= siluf_(z);
    ybf[(size_t)l * DI + d] = f2bf(y);
  }
}

__global__ __launch_bounds__(256)
void k_add_rms_quant(const float* __restrict__ x, const float* __restrict__ h,
                     const float* __restrict__ w, float* __restrict__ x1,
                     signed char* __restrict__ xq, float* __restrict__ rs, int RL)
{
  __shared__ float sred[256];
  int row = blockIdx.x, tid = threadIdx.x;
  const float* xr = x + (size_t)row * RL;
  const float* hr = h + (size_t)row * RL;
  float ss = 0.f;
  for (int c = tid; c < RL; c += 256) { float v = xr[c] + hr[c]; ss += v * v; }
  ss = blkSum(ss, sred);
  float inv = rsqrtf(ss / (float)RL + 1e-6f);
  float* x1r = x1 + (size_t)row * RL;
  float mx = 0.f;
  for (int c = tid; c < RL; c += 256) {
    float v = (xr[c] + hr[c]) * inv * w[c];
    x1r[c] = v;
    mx = fmaxf(mx, fabsf(v));
  }
  mx = blkMax(mx, sred);
  float s = 127.f / fmaxf(mx, 1e-5f);
  if (tid == 0) rs[row] = 1.f / s;
  signed char* qr = xq + (size_t)row * RL;
  for (int c = tid; c < RL; c += 256) {
    float q = rintf(x1r[c] * s);
    q = fminf(fmaxf(q, -128.f), 127.f);
    qr[c] = (signed char)(int)q;
  }
}

__global__ __launch_bounds__(256)
void k_rowquant(const float* __restrict__ f, signed char* __restrict__ fq,
                float* __restrict__ rs, int RL)
{
  __shared__ float sred[256];
  int row = blockIdx.x, tid = threadIdx.x;
  const float* fr = f + (size_t)row * RL;
  float mx = 0.f;
  for (int c = tid; c < RL; c += 256) mx = fmaxf(mx, fabsf(fr[c]));
  mx = blkMax(mx, sred);
  float s = 127.f / fmaxf(mx, 1e-5f);
  if (tid == 0) rs[row] = 1.f / s;
  signed char* qr = fq + (size_t)row * RL;
  for (int c = tid; c < RL; c += 256) {
    float q = rintf(fr[c] * s);
    q = fminf(fmaxf(q, -128.f), 127.f);
    qr[c] = (signed char)(int)q;
  }
}

__global__ __launch_bounds__(256)
void k_add_rms_out(const float* __restrict__ a, const float* __restrict__ b,
                   const float* __restrict__ w, float* __restrict__ out, int RL)
{
  __shared__ float sred[256];
  int row = blockIdx.x, tid = threadIdx.x;
  const float* ar = a + (size_t)row * RL;
  const float* br = b + (size_t)row * RL;
  float ss = 0.f;
  for (int c = tid; c < RL; c += 256) { float v = ar[c] + br[c]; ss += v * v; }
  ss = blkSum(ss, sred);
  float inv = rsqrtf(ss / (float)RL + 1e-6f);
  float* orow = out + (size_t)row * RL;
  for (int c = tid; c < RL; c += 256) orow[c] = (ar[c] + br[c]) * inv * w[c];
}

__global__ __launch_bounds__(256)
void k_absmean_part(const float* __restrict__ w, float* __restrict__ part, int n) {
  __shared__ float sred[256];
  float s = 0.f;
  for (int i = blockIdx.x * 256 + threadIdx.x; i < n; i += 256 * gridDim.x) s += fabsf(w[i]);
  s = blkSum(s, sred);
  if (threadIdx.x == 0) part[blockIdx.x] = s;
}
__global__ __launch_bounds__(256)
void k_absmean_fin(const float* __restrict__ part, int nb, float* __restrict__ g, float invn) {
  __shared__ float sred[256];
  float s = 0.f;
  for (int i = threadIdx.x; i < nb; i += 256) s += part[i];
  s = blkSum(s, sred);
  if (threadIdx.x == 0) g[0] = fmaxf(s * invn, 1e-5f);
}
__global__ void k_wquant(const float* __restrict__ w, const float* __restrict__ g,
                         signed char* __restrict__ wq, int n) {
  int i = blockIdx.x * blockDim.x + threadIdx.x;
  if (i >= n) return;
  float q = rintf(w[i] / g[0]);
  q = fminf(fmaxf(q, -1.f), 1.f);
  wq[i] = (signed char)(int)q;
}

extern "C" void kernel_launch(void* const* d_in, const int* in_sizes, int n_in,
                              void* d_out, int out_size, void* d_ws, size_t ws_size,
                              hipStream_t stream)
{
  (void)in_sizes; (void)n_in; (void)out_size; (void)ws_size;
  const float* x      = (const float*)d_in[0];
  const float* w_in   = (const float*)d_in[1];
  const float* conv_w = (const float*)d_in[2];
  const float* conv_b = (const float*)d_in[3];
  const float* w_xprj = (const float*)d_in[4];
  const float* w_dt   = (const float*)d_in[5];
  const float* b_dt   = (const float*)d_in[6];
  const float* A_log  = (const float*)d_in[7];
  const float* Dp     = (const float*)d_in[8];
  const float* w_out  = (const float*)d_in[9];
  const float* n1w    = (const float*)d_in[10];
  const float* n2w    = (const float*)d_in[11];
  const float* fw1    = (const float*)d_in[12];
  const float* fb1    = (const float*)d_in[13];
  const float* fw2    = (const float*)d_in[14];
  const float* fb2    = (const float*)d_in[15];
  float* out = (float*)d_out;

  char* ws = (char*)d_ws;
  size_t cur = 0;
  auto take = [&](size_t bytes) -> char* {
    char* p = ws + cur;
    cur = (cur + bytes + 255) & ~(size_t)255;
    return p;
  };
  float*          xz    = (float*)         take((size_t)LSEQ * XZW * 4);
  unsigned short* xbf   = (unsigned short*)take((size_t)LSEQ * DM * 2);
  unsigned short* winbf = (unsigned short*)take((size_t)XZW * DM * 2);
  float*          u     = (float*)         take((size_t)LSEQ * DI * 4);
  unsigned short* ubf   = (unsigned short*)take((size_t)LSEQ * DI * 2);
  unsigned short* wxbf  = (unsigned short*)take((size_t)DBLW * DI * 2);
  float*          dbl   = (float*)         take((size_t)LSEQ * DBLW * 4);
  unsigned short* dblbf = (unsigned short*)take((size_t)LSEQ * DBLW * 2);
  unsigned short* wdtbf = (unsigned short*)take((size_t)DI * DTR * 2);
  float*          delta = (float*)         take((size_t)LSEQ * DI * 4);
  unsigned short* ybf   = (unsigned short*)take((size_t)LSEQ * DI * 2);
  unsigned short* wobf  = (unsigned short*)take((size_t)DM * DI * 2);
  float*          hbuf  = (float*)         take((size_t)LSEQ * DM * 4);
  float*          x1    = (float*)         take((size_t)LSEQ * DM * 4);
  signed char*    x1q   = (signed char*)   take((size_t)LSEQ * DM);
  float*          rs1   = (float*)         take((size_t)LSEQ * 4);
  signed char*    w1q   = (signed char*)   take((size_t)DFF * DM);
  signed char*    w2q   = (signed char*)   take((size_t)DM * DFF);
  signed char*    fq    = (signed char*)   take((size_t)LSEQ * DFF);
  float*          rs2   = (float*)         take((size_t)LSEQ * 4);
  float*          f2    = (float*)         take((size_t)LSEQ * DM * 4);
  float*          part  = (float*)         take(256 * 4);
  float*          g1    = (float*)         take(256);
  float*          g2    = (float*)         take(256);
  float*          fbuf  = xz;

  dim3 blk(256);

  k_cvt_bf16<<<dim3((LSEQ * DM + 255) / 256), blk, 0, stream>>>(x, xbf, LSEQ * DM);
  k_cvt_bf16<<<dim3((XZW * DM + 255) / 256), blk, 0, stream>>>(w_in, winbf, XZW * DM);
  k_cvt_bf16<<<dim3((DBLW * DI + 255) / 256), blk, 0, stream>>>(w_xprj, wxbf, DBLW * DI);
  k_cvt_bf16<<<dim3((DI * DTR + 255) / 256), blk, 0, stream>>>(w_dt, wdtbf, DI * DTR);
  k_cvt_bf16<<<dim3((DM * DI + 255) / 256), blk, 0, stream>>>(w_out, wobf, DM * DI);

  k_gemm_bf16<EPI_NONE><<<dim3(XZW / BN, LSEQ / BM), blk, 0, stream>>>(
      xbf, DM, winbf, DM, xz, XZW, nullptr, nullptr, LSEQ, XZW, DM);

  k_conv_silu<<<dim3(LSEQ * DI / 256), blk, 0, stream>>>(xz, conv_w, conv_b, u, ubf);

  k_gemm_bf16<EPI_STORE_BF16><<<dim3(1, LSEQ / BM), blk, 0, stream>>>(
      ubf, DI, wxbf, DI, dbl, DBLW, dblbf, nullptr, LSEQ, DBLW, DI);

  k_gemm_bf16<EPI_SOFTPLUS_BIAS><<<dim3(DI / BN, LSEQ / BM), blk, 0, stream>>>(
      dblbf, DBLW, wdtbf, DTR, delta, DI, nullptr, b_dt, LSEQ, DI, DTR);

  k_scan<<<dim3(DI / 256), blk, 0, stream>>>(delta, u, dbl, A_log, Dp, xz, ybf);

  k_gemm_bf16<EPI_NONE><<<dim3(DM / BN, LSEQ / BM), blk, 0, stream>>>(
      ybf, DI, wobf, DI, hbuf, DM, nullptr, nullptr, LSEQ, DM, DI);

  k_add_rms_quant<<<dim3(LSEQ), blk, 0, stream>>>(x, hbuf, n1w, x1, x1q, rs1, DM);

  k_absmean_part<<<dim3(256), blk, 0, stream>>>(fw1, part, DFF * DM);
  k_absmean_fin<<<dim3(1), blk, 0, stream>>>(part, 256, g1, 1.f / (float)(DFF * DM));
  k_wquant<<<dim3((DFF * DM + 255) / 256), blk, 0, stream>>>(fw1, g1, w1q, DFF * DM);

  k_gemm_i8<1><<<dim3(DFF / BN, LSEQ / BM), blk, 0, stream>>>(
      x1q, DM, w1q, DM, rs1, g1, fb1, fbuf, DFF, LSEQ, DFF, DM);

  k_rowquant<<<dim3(LSEQ), blk, 0, stream>>>(fbuf, fq, rs2, DFF);

  k_absmean_part<<<dim3(256), blk, 0, stream>>>(fw2, part, DM * DFF);
  k_absmean_fin<<<dim3(1), blk, 0, stream>>>(part, 256, g2, 1.f / (float)(DM * DFF));
  k_wquant<<<dim3((DM * DFF + 255) / 256), blk, 0, stream>>>(fw2, g2, w2q, DM * DFF);

  k_gemm_i8<0><<<dim3(DM / BN, LSEQ / BM), blk, 0, stream>>>(
      fq, DFF, w2q, DFF, rs2, g2, fb2, f2, DM, LSEQ, DM, DFF);

  k_add_rms_out<<<dim3(LSEQ), blk, 0, stream>>>(x1, f2, n2w, out, DM);
}
